// FeedForwardSNN_81097572483821
// MI455X (gfx1250) — matrix operand order repaired and checked by reading, not compiled, not run on hardware
//
#include <hip/hip_runtime.h>

typedef __attribute__((ext_vector_type(16))) _Float16 v16h;
typedef __attribute__((ext_vector_type(8)))  _Float16 v8h;
typedef __attribute__((ext_vector_type(4)))  _Float16 v4h;
typedef __attribute__((ext_vector_type(8)))  float    v8f;

#define T_STEPS 128
#define BATCH   1024
#define D_IN    96
#define H_DIM   128
#define BIN     50
#define NSTEPS  7

// ---------------------------------------------------------------------------
// Kernel A: PY[j][b][h] = sum_{f<j} (x[f] @ W_in^T)[b][h],  j = 0..128
// Each wave owns a 16x16 (batch x h) tile and walks t sequentially,
// accumulating the prefix directly in the WMMA C/D accumulator.
// ---------------------------------------------------------------------------
__global__ __launch_bounds__(256) void snn_prefix_gemm(
    const float* __restrict__ x,      // (128,1024,96)
    const float* __restrict__ Win,    // (128,96)
    float* __restrict__ PY)           // (129,1024,128)
{
  const int lane = threadIdx.x & 31;
  const int ln   = lane & 15;
  const int hf   = lane >> 4;
  const int gw   = blockIdx.x * 8 + (threadIdx.x >> 5);  // 0..511
  const int b0   = (gw >> 3) * 16;                       // batch tile base
  const int h0   = (gw & 7) * 16;                        // h tile base

  // B fragments (t-invariant): W_in rows, n = h0+ln, k = kt*32 + 16*hf + j
  v16h Bf[3];
#pragma unroll
  for (int kt = 0; kt < 3; ++kt) {
    const float* wp = Win + (h0 + ln) * D_IN + kt * 32 + 16 * hf;
    v16h f;
#pragma unroll
    for (int j = 0; j < 16; ++j) f[j] = (_Float16)wp[j];
    Bf[kt] = f;
  }

  // PY[0] = 0
  {
    float* p0 = PY + (size_t)b0 * H_DIM + h0;
#pragma unroll
    for (int r = 0; r < 8; ++r) p0[(r + 8 * hf) * H_DIM + ln] = 0.0f;
  }

  v8f acc = {};  // running prefix tile
  for (int t = 0; t < T_STEPS; ++t) {
    const float* xr = x + ((size_t)t * BATCH + (b0 + ln)) * D_IN;
#pragma unroll
    for (int kt = 0; kt < 3; ++kt) {
      // A layout: lane holds M=ln; k = k0 + 16*hf + {0..15}
      const float* ap = xr + kt * 32 + 16 * hf;
      float tmp[16];
      *(float4*)(tmp + 0)  = *(const float4*)(ap + 0);
      *(float4*)(tmp + 4)  = *(const float4*)(ap + 4);
      *(float4*)(tmp + 8)  = *(const float4*)(ap + 8);
      *(float4*)(tmp + 12) = *(const float4*)(ap + 12);
      v16h A;
#pragma unroll
      for (int j = 0; j < 16; ++j) A[j] = (_Float16)tmp[j];
      acc = __builtin_amdgcn_wmma_f32_16x16x32_f16(false, A, false, Bf[kt],
                                                   (short)0, acc, false, false);
    }
    float* po = PY + ((size_t)(t + 1) * BATCH + b0) * H_DIM + h0;
#pragma unroll
    for (int r = 0; r < 8; ++r) po[(r + 8 * hf) * H_DIM + ln] = acc[r];
  }
}

// Layer-1 LIF update body; GET_VLO selects cached vs freshly-loaded PY[jlo].
#define LAYER1_BODY(GET_VLO)                                              \
  _Pragma("unroll")                                                       \
  for (int nt = 0; nt < 8; ++nt) {                                        \
    _Pragma("unroll")                                                     \
    for (int r = 0; r < 8; ++r) {                                         \
      const int lrow = m0 + r + 8 * hf;                                   \
      const int col  = nt * 16 + ln;                                      \
      const size_t off = (size_t)(bbase + lrow) * H_DIM + col;            \
      const float vhi = Phi[off];                                         \
      const float vlo = (GET_VLO);                                        \
      curHi[nt][r] = vhi;                                                 \
      const float cur = vhi - vlo;                                        \
      float m = mem1[nt][r];                                              \
      const float rst = (m > 0.5f) ? 1.0f : 0.0f;                         \
      m = (0.9f * m + cur) * (1.0f - rst);                                \
      mem1[nt][r] = m;                                                    \
      sS1[lrow * H_DIM + col] = (_Float16)((m > 0.5f) ? 1.0f : 0.0f);     \
    }                                                                     \
  }

// ---------------------------------------------------------------------------
// Kernel B: sequential 7-step SNN per t. Grid = 128 t * 8 batch-tiles.
// Wave w owns batch rows [btile*128 + 16w, +16) x all H.
// mem1/mem2 live in WMMA C-layout accumulators; spikes round-trip through LDS
// (each wave reads only its own rows -> no per-step barrier needed).
// PY windows telescope: PY[jhi] of step s == PY[jlo] of step s+1 -> cached.
// ---------------------------------------------------------------------------
__global__ __launch_bounds__(256) void snn_seq(
    const float* __restrict__ PY,     // (129,1024,128)
    const float* __restrict__ Wh0,    // (128,128)
    const float* __restrict__ Wout,   // (2,128)
    const float* __restrict__ vx,
    const float* __restrict__ vy,
    float* __restrict__ out)          // (128,1024,2)
{
  __shared__ _Float16 sWh[H_DIM * H_DIM];   // 32 KB, W_h0 as f16 [n][k]
  __shared__ _Float16 sS1[H_DIM * H_DIM];   // 32 KB, spk1 [m][k]

  const int tid  = threadIdx.x;
  const int lane = tid & 31;
  const int ln   = lane & 15;
  const int hf   = lane >> 4;
  const int w    = tid >> 5;                // wave 0..7
  const int t     = blockIdx.x >> 3;
  const int bbase = (blockIdx.x & 7) * 128; // batch base of block
  const int m0    = w * 16;                 // wave row base within block

  for (int i = tid * 4; i < H_DIM * H_DIM; i += 256 * 4) {
    float4 v = *(const float4*)(Wh0 + i);
    v4h h;
    h[0] = (_Float16)v.x; h[1] = (_Float16)v.y;
    h[2] = (_Float16)v.z; h[3] = (_Float16)v.w;
    *(v4h*)(&sWh[i]) = h;
  }
  __syncthreads();

  float wo0[8], wo1[8];
#pragma unroll
  for (int nt = 0; nt < 8; ++nt) {
    wo0[nt] = Wout[nt * 16 + ln];
    wo1[nt] = Wout[H_DIM + nt * 16 + ln];
  }

  v8f mem1[8], mem2[8];
#pragma unroll
  for (int nt = 0; nt < 8; ++nt) { mem1[nt] = (v8f){}; mem2[nt] = (v8f){}; }
  float curHi[8][8];   // cached PY[jhi] tile values (telescoping windows)
  float mo0[8] = {0, 0, 0, 0, 0, 0, 0, 0};
  float mo1[8] = {0, 0, 0, 0, 0, 0, 0, 0};
  int jprev = -1;

  for (int s = 0; s < NSTEPS; ++s) {
    // frame window [lo,hi] inclusive into x; empty -> zero
    int lo, hiF;
    if (t >= BIN - 1) { lo = t - (BIN - 1) + 7 * s; hiF = lo + 6; }
    else {
      lo = 7 * s - 1; if (lo < 0) lo = 0;
      hiF = 7 * s + 5; if (hiF > t) hiF = t;
    }
    int jhi = hiF + 1, jlo = lo;
    if (hiF < lo) { jhi = 0; jlo = 0; }
    const float* Phi = PY + (size_t)jhi * (BATCH * H_DIM);
    const float* Plo = PY + (size_t)jlo * (BATCH * H_DIM);

    // ---- layer 1: mem1 LIF update, spk1 -> LDS (f16) ----
    if (jlo == jprev) { LAYER1_BODY(curHi[nt][r]) }
    else              { LAYER1_BODY(Plo[off]) }
    jprev = jhi;

    // ---- layer 2 reset (from OLD mem2), pre-scale, pack reset bits ----
    unsigned rstMask[2] = {0u, 0u};
#pragma unroll
    for (int nt = 0; nt < 8; ++nt) {
#pragma unroll
      for (int r = 0; r < 8; ++r) {
        const int bit = nt * 8 + r;
        if (mem2[nt][r] > 0.5f) rstMask[bit >> 5] |= (1u << (bit & 31));
        mem2[nt][r] *= 0.9f;
      }
    }

    // ---- GEMM2: mem2 += spk1 @ W_h0^T  (4 K-steps x 8 N-tiles of WMMA) ----
#pragma unroll
    for (int kt = 0; kt < 4; ++kt) {
      const _Float16* ap = &sS1[(m0 + ln) * H_DIM + kt * 32 + 16 * hf];
      v8h a0 = *(const v8h*)(ap);
      v8h a1 = *(const v8h*)(ap + 8);
      v16h A = __builtin_shufflevector(a0, a1, 0, 1, 2, 3, 4, 5, 6, 7,
                                       8, 9, 10, 11, 12, 13, 14, 15);
#pragma unroll
      for (int nt = 0; nt < 8; ++nt) {
        const _Float16* bp = &sWh[(nt * 16 + ln) * H_DIM + kt * 32 + 16 * hf];
        v16h B = *(const v16h*)(bp);
        mem2[nt] = __builtin_amdgcn_wmma_f32_16x16x32_f16(
            false, A, false, B, (short)0, mem2[nt], false, false);
      }
    }

    // ---- apply reset, spk2, output-layer partial dots (O=2) ----
    float p0[8] = {0, 0, 0, 0, 0, 0, 0, 0};
    float p1[8] = {0, 0, 0, 0, 0, 0, 0, 0};
#pragma unroll
    for (int nt = 0; nt < 8; ++nt) {
#pragma unroll
      for (int r = 0; r < 8; ++r) {
        const int bit = nt * 8 + r;
        float m = mem2[nt][r];
        if (rstMask[bit >> 5] & (1u << (bit & 31))) m = 0.0f;
        mem2[nt][r] = m;
        if (m > 0.5f) { p0[r] += wo0[nt]; p1[r] += wo1[nt]; }
      }
    }
    // reduce over the 16 lanes of each half (n dimension)
#pragma unroll
    for (int msk = 1; msk < 16; msk <<= 1) {
#pragma unroll
      for (int r = 0; r < 8; ++r) {
        p0[r] += __shfl_xor(p0[r], msk, 32);
        p1[r] += __shfl_xor(p1[r], msk, 32);
      }
    }
#pragma unroll
    for (int r = 0; r < 8; ++r) {
      mo0[r] = 0.9f * mo0[r] + p0[r];
      mo1[r] = 0.9f * mo1[r] + p1[r];
    }
  }

  // ---- write preds: out[t][b][:] = mem_out * [v_x, v_y] ----
  const float vxs = vx[0], vys = vy[0];
  if (ln == 0) {
#pragma unroll
    for (int r = 0; r < 8; ++r) {
      const int b = bbase + m0 + r + 8 * hf;
      float2 o = make_float2(mo0[r] * vxs, mo1[r] * vys);
      *(float2*)(&out[((size_t)t * BATCH + b) * 2]) = o;
    }
  }
}

extern "C" void kernel_launch(void* const* d_in, const int* in_sizes, int n_in,
                              void* d_out, int out_size, void* d_ws, size_t ws_size,
                              hipStream_t stream) {
  const float* x    = (const float*)d_in[0];   // (128,1024,96)
  const float* Win  = (const float*)d_in[1];   // (128,96)
  const float* Wh0  = (const float*)d_in[2];   // (128,128)
  const float* Wout = (const float*)d_in[3];   // (2,128)
  const float* vx   = (const float*)d_in[4];   // (1,)
  const float* vy   = (const float*)d_in[5];   // (1,)
  float* PY = (float*)d_ws;                    // (129,1024,128) f32 = 67.6 MB

  snn_prefix_gemm<<<dim3(64), dim3(256), 0, stream>>>(x, Win, PY);
  snn_seq<<<dim3(1024), dim3(256), 0, stream>>>(PY, Wh0, Wout, vx, vy,
                                                (float*)d_out);
}
